// WindowAttention_73847667687716
// MI455X (gfx1250) — matrix operand order repaired and checked by reading, not compiled, not run on hardware
//
#include <hip/hip_runtime.h>

// ---------------------------------------------------------------------------
// Swin window attention for MI455X (gfx1250), wave32 + v_wmma_f32_16x16x32_bf16
//   B=4096 windows, N=49 tokens, DIM=512, HEADS=16, HEAD_DIM=32
// Pipeline: bias-table expand -> QKV GEMM (bf16) -> fused attention -> proj GEMM
// ---------------------------------------------------------------------------

typedef __attribute__((ext_vector_type(16))) __bf16 bf16x16;
typedef __attribute__((ext_vector_type(8)))  float  floatx8;
typedef unsigned short u16;
typedef unsigned int   u32;

union FragA {                 // 32 bytes = 16 bf16 = one WMMA A/B fragment
  uint4   u[2];
  bf16x16 v;
};

__device__ __forceinline__ u16 f32_to_bf16_bits(float f) {
  return __builtin_bit_cast(u16, (__bf16)f);
}

__device__ __forceinline__ void cvt_store8(u16* d, float4 a, float4 b) {
  d[0] = f32_to_bf16_bits(a.x); d[1] = f32_to_bf16_bits(a.y);
  d[2] = f32_to_bf16_bits(a.z); d[3] = f32_to_bf16_bits(a.w);
  d[4] = f32_to_bf16_bits(b.x); d[5] = f32_to_bf16_bits(b.y);
  d[6] = f32_to_bf16_bits(b.z); d[7] = f32_to_bf16_bits(b.w);
}

// ---------------------------------------------------------------------------
// Expand rel-pos bias table into padded, mask-folded bias_full[16][64][64].
//   n >= 49  -> -1e30 (padding-column mask, kills softmax contribution)
//   m >= 49  -> 0     (rows never stored)
// ---------------------------------------------------------------------------
__global__ __launch_bounds__(256)
void build_bias_full(const float* __restrict__ tbl,   // [169, 16]
                     float* __restrict__ bias_full)   // [16, 64, 64]
{
  const int i = blockIdx.x * 256 + threadIdx.x;       // h*4096 + m*64 + n
  const int n = i & 63;
  const int m = (i >> 6) & 63;
  const int h = i >> 12;
  float v;
  if (n >= 49) {
    v = -1e30f;
  } else if (m >= 49) {
    v = 0.0f;
  } else {
    const int rm = m / 7, cm = m % 7;
    const int rn = n / 7, cn = n % 7;
    v = tbl[((rm - rn + 6) * 13 + (cm - cn + 6)) * 16 + h];
  }
  bias_full[i] = v;
}

// ---------------------------------------------------------------------------
// Generic bf16 WMMA GEMM:  C[M,N] = A[M,K] * B[K,N] + bias[N]
//   Block tile 128x128x32, 8 waves (4Mx2N), wave tile 32x64 = 2x4 WMMA tiles
// ---------------------------------------------------------------------------
template<bool A_F32, bool OUT_BF16>
__global__ __launch_bounds__(256)
void gemm_bf16_wmma(const void*  __restrict__ Ap,
                    const float* __restrict__ Bp,
                    const float* __restrict__ bias,
                    void*        __restrict__ Cp,
                    int M, int N, int K)
{
  constexpr int BM = 128, BN = 128, BK = 32;
  constexpr int LDL = BK + 8;              // LDS row pitch (pad vs bank conflicts)
  __shared__ u16 As[BM][LDL];              // [row][k]
  __shared__ u16 BsT[BN][LDL];             // [col][k]  (transposed stage)

  const int tid  = threadIdx.x;
  const int lane = tid & 31;
  const int l16  = lane & 15;
  const bool hi  = lane >= 16;
  const int wave = tid >> 5;
  const int wm   = wave >> 1;              // 0..3 -> 32-row strip
  const int wn   = wave & 1;               // 0..1 -> 64-col strip
  const long bm  = (long)blockIdx.y * BM;
  const long bn  = (long)blockIdx.x * BN;

  floatx8 acc[2][4] = {};

  const int arow = tid >> 1;               // 0..127 (2 threads per A row)
  const int aseg = (tid & 1) << 4;         // 0 / 16 elems
  const int brow = tid >> 3;               // 0..31  (8 threads per B row)
  const int bseg = (tid & 7) << 4;         // 0..112 step 16

  for (int kk = 0; kk < K; kk += BK) {
    // ---- stage A tile (convert f32->bf16 if needed) ----
    if (A_F32) {
      const float* A = (const float*)Ap;
      const float4* s = (const float4*)(A + (size_t)(bm + arow) * K + kk + aseg);
      float4 f0 = s[0], f1 = s[1], f2 = s[2], f3 = s[3];
      u16* d = &As[arow][aseg];
      cvt_store8(d, f0, f1);
      cvt_store8(d + 8, f2, f3);
    } else {
      const u16* A = (const u16*)Ap;
      const uint4* s = (const uint4*)(A + (size_t)(bm + arow) * K + kk + aseg);
      uint4* d = (uint4*)&As[arow][aseg];
      d[0] = s[0]; d[1] = s[1];
    }
    // ---- stage B tile transposed: 16 cols per thread of one K-row ----
    {
      const float4* s = (const float4*)(Bp + (size_t)(kk + brow) * N + bn + bseg);
      float4 f0 = s[0], f1 = s[1], f2 = s[2], f3 = s[3];
      float vals[16] = { f0.x, f0.y, f0.z, f0.w, f1.x, f1.y, f1.z, f1.w,
                         f2.x, f2.y, f2.z, f2.w, f3.x, f3.y, f3.z, f3.w };
      #pragma unroll
      for (int i = 0; i < 16; ++i)
        BsT[bseg + i][brow] = f32_to_bf16_bits(vals[i]);
    }
    __syncthreads();

    // ---- fragments + WMMA ----
    FragA fa[2], fb[4];
    #pragma unroll
    for (int sm = 0; sm < 2; ++sm) {
      const u16* p = &As[wm * 32 + sm * 16 + l16][hi ? 8 : 0];
      fa[sm].u[0] = *(const uint4*)p;          // K 0..7  / 8..15
      fa[sm].u[1] = *(const uint4*)(p + 16);   // K 16..23 / 24..31
    }
    #pragma unroll
    for (int sn = 0; sn < 4; ++sn) {
      const u16* p = &BsT[wn * 64 + sn * 16 + l16][hi ? 8 : 0];
      fb[sn].u[0] = *(const uint4*)p;          // K 0..7  / 8..15
      fb[sn].u[1] = *(const uint4*)(p + 16);   // K 16..23 / 24..31
    }
    #pragma unroll
    for (int sm = 0; sm < 2; ++sm)
      #pragma unroll
      for (int sn = 0; sn < 4; ++sn)
        acc[sm][sn] = __builtin_amdgcn_wmma_f32_16x16x32_bf16(
            false, fa[sm].v, false, fb[sn].v, (short)0, acc[sm][sn],
            false, false);
    __syncthreads();
  }

  // ---- epilogue: bias + store ----
  #pragma unroll
  for (int sm = 0; sm < 2; ++sm) {
    #pragma unroll
    for (int sn = 0; sn < 4; ++sn) {
      const long col = bn + wn * 64 + sn * 16 + l16;
      const float bv = bias[col];
      #pragma unroll
      for (int r = 0; r < 8; ++r) {
        const long row = bm + wm * 32 + sm * 16 + r + (hi ? 8 : 0);
        const float v = acc[sm][sn][r] + bv;
        if (OUT_BF16)
          ((u16*)Cp)[(size_t)row * N + col] = f32_to_bf16_bits(v);
        else
          ((float*)Cp)[(size_t)row * N + col] = v;
      }
    }
  }
}

// ---------------------------------------------------------------------------
// Fused window attention: one wave per (window b, head h).
//   qkv       : [B*49, 1536] bf16   (q | k | v, per head 32 cols)
//   bias_full : [16, 64, 64] f32, mask folded in, single-vaddr clause loads
//   aout      : [B*49, 512] bf16    attention output (input to proj GEMM)
// LDS regions are wave-private -> s_wait_dscnt 0 instead of block barriers.
// ---------------------------------------------------------------------------
__global__ __launch_bounds__(128)
void window_attn_wmma(const u16*  __restrict__ qkv,
                      const float* __restrict__ bias_full,
                      u16*        __restrict__ aout)
{
  constexpr int   NT    = 49;
  constexpr float SCALE = 0.17677669529663687f;   // 32^-0.5

  __shared__ u16 Pl[4][64][72];   // softmax(P) bf16, per wave
  __shared__ u16 Vt[4][32][72];   // V^T bf16 [head_dim][64], per wave

  const int  lane = threadIdx.x & 31;
  const int  wave = threadIdx.x >> 5;
  const int  l16  = lane & 15;
  const bool hi   = lane >= 16;
  const int  pair = blockIdx.x * 4 + wave;
  const int  b    = pair >> 4;
  const int  h    = pair & 15;

  const size_t base = (size_t)b * 49 * 1536 + (size_t)h * 32;
  const u16* qb = qkv + base;
  const u16* kb = qb + 512;
  const u16* vb = qb + 1024;

  // ---- Q fragments (A layout: row per lane, K split across half-waves) ----
  FragA qa[4];
  #pragma unroll
  for (int mt = 0; mt < 4; ++mt) {
    int row = mt * 16 + l16; if (row > 48) row = 48;   // clamp pad rows
    const u16* p = qb + (size_t)row * 1536 + (hi ? 8 : 0);
    qa[mt].u[0] = *(const uint4*)p;
    qa[mt].u[1] = *(const uint4*)(p + 16);
  }
  // ---- K fragments (B layout: K^T column n == K row n, contiguous) ----
  FragA kf[4];
  #pragma unroll
  for (int nt = 0; nt < 4; ++nt) {
    int col = nt * 16 + l16; if (col > 48) col = 48;
    const u16* p = kb + (size_t)col * 1536 + (hi ? 8 : 0);
    kf[nt].u[0] = *(const uint4*)p;
    kf[nt].u[1] = *(const uint4*)(p + 16);
  }

  // ---- V^T -> LDS early, packed: lane owns rows k0,k0+1 which form one
  //      aligned dword in VT[c][k] -> 32 b32 stores, 1 pack op per dword ----
  u16 (*VT)[72] = Vt[wave];
  {
    const int k0 = lane * 2;
    const int r0 = (k0     > 48) ? 48 : k0;
    const int r1 = (k0 + 1 > 48) ? 48 : k0 + 1;
    const uint4* s0 = (const uint4*)(vb + (size_t)r0 * 1536);
    const uint4* s1 = (const uint4*)(vb + (size_t)r1 * 1536);
    union { uint4 u[4]; u16 e[32]; } t0, t1;
    t0.u[0] = s0[0]; t0.u[1] = s0[1]; t0.u[2] = s0[2]; t0.u[3] = s0[3];
    t1.u[0] = s1[0]; t1.u[1] = s1[1]; t1.u[2] = s1[2]; t1.u[3] = s1[3];
    #pragma unroll
    for (int c = 0; c < 32; ++c) {
      const u32 pk = (u32)t0.e[c] | ((u32)t1.e[c] << 16);
      *(u32*)&VT[c][k0] = pk;
    }
  }

  // ---- logits = Q K^T : 16 WMMAs, K = head_dim = 32 exactly ----
  floatx8 logit[4][4];
  #pragma unroll
  for (int mt = 0; mt < 4; ++mt)
    #pragma unroll
    for (int nt = 0; nt < 4; ++nt) {
      floatx8 z = {};
      logit[mt][nt] = __builtin_amdgcn_wmma_f32_16x16x32_bf16(
          false, qa[mt].v, false, kf[nt].v, (short)0, z, false, false);
    }

  // ---- fused: scale + bias + row softmax + bf16 store to LDS ----
  // logit elements are read exactly once (no accumulator repacking).
  // element (m, n): m = mt*16 + r + hi*8, n = nt*16 + l16
  const float* bh = bias_full + ((size_t)h << 12) + (hi ? 8 * 64 : 0) + l16;
  u16 (*P)[72] = Pl[wave];
  #pragma unroll
  for (int mt = 0; mt < 4; ++mt) {
    #pragma unroll
    for (int r = 0; r < 8; ++r) {
      const int boff = mt * 1024 + r * 64;
      float x0 = logit[mt][0][r] * SCALE + bh[boff];
      float x1 = logit[mt][1][r] * SCALE + bh[boff + 16];
      float x2 = logit[mt][2][r] * SCALE + bh[boff + 32];
      float x3 = logit[mt][3][r] * SCALE + bh[boff + 48];
      float mx = fmaxf(fmaxf(x0, x1), fmaxf(x2, x3));
      mx = fmaxf(mx, __shfl_xor(mx, 1, 32));
      mx = fmaxf(mx, __shfl_xor(mx, 2, 32));
      mx = fmaxf(mx, __shfl_xor(mx, 4, 32));
      mx = fmaxf(mx, __shfl_xor(mx, 8, 32));
      float e0 = __expf(x0 - mx), e1 = __expf(x1 - mx);
      float e2 = __expf(x2 - mx), e3 = __expf(x3 - mx);
      float s = e0 + e1 + e2 + e3;
      s += __shfl_xor(s, 1, 32);
      s += __shfl_xor(s, 2, 32);
      s += __shfl_xor(s, 4, 32);
      s += __shfl_xor(s, 8, 32);
      const float inv = 1.0f / s;
      const int m = mt * 16 + r + (hi ? 8 : 0);
      u16* pr = &P[m][l16];
      pr[0]  = f32_to_bf16_bits(e0 * inv);
      pr[16] = f32_to_bf16_bits(e1 * inv);
      pr[32] = f32_to_bf16_bits(e2 * inv);
      pr[48] = f32_to_bf16_bits(e3 * inv);
    }
  }

  // LDS regions are wave-private: wait for this wave's DS stores only
  // (CDNA5 split counters), no block-wide barrier needed.
  asm volatile("s_wait_dscnt 0" ::: "memory");

  // ---- out = P @ V : 4 row tiles x 2 col tiles x 2 K-steps = 16 WMMAs ----
  floatx8 oacc[4][2] = {};
  #pragma unroll
  for (int mt = 0; mt < 4; ++mt)
    #pragma unroll
    for (int kt = 0; kt < 2; ++kt) {
      FragA pa;
      const u16* pp = &P[mt * 16 + l16][kt * 32 + (hi ? 8 : 0)];
      pa.u[0] = *(const uint4*)pp;
      pa.u[1] = *(const uint4*)(pp + 16);
      #pragma unroll
      for (int ct = 0; ct < 2; ++ct) {
        FragA vf;
        const u16* vp = &VT[ct * 16 + l16][kt * 32 + (hi ? 8 : 0)];
        vf.u[0] = *(const uint4*)vp;
        vf.u[1] = *(const uint4*)(vp + 16);
        oacc[mt][ct] = __builtin_amdgcn_wmma_f32_16x16x32_bf16(
            false, pa.v, false, vf.v, (short)0, oacc[mt][ct], false, false);
      }
    }

  // ---- store rows < 49 as bf16 into [B*49, 512] (head-major columns) ----
  #pragma unroll
  for (int mt = 0; mt < 4; ++mt)
    #pragma unroll
    for (int ct = 0; ct < 2; ++ct) {
      const int c = h * 32 + ct * 16 + l16;
      #pragma unroll
      for (int r = 0; r < 8; ++r) {
        const int m = mt * 16 + r + (hi ? 8 : 0);
        if (m < NT)
          aout[((size_t)b * 49 + m) * 512 + c] =
              f32_to_bf16_bits(oacc[mt][ct][r]);
      }
    }
}

// ---------------------------------------------------------------------------
// Launch: bias expand -> QKV GEMM -> fused attention -> proj GEMM
// ---------------------------------------------------------------------------
extern "C" void kernel_launch(void* const* d_in, const int* in_sizes, int n_in,
                              void* d_out, int out_size, void* d_ws, size_t ws_size,
                              hipStream_t stream)
{
  (void)in_sizes; (void)n_in; (void)out_size; (void)ws_size;

  const float* x       = (const float*)d_in[0];  // [4096*49, 512]
  const float* qkv_w   = (const float*)d_in[1];  // [512, 1536]
  const float* qkv_b   = (const float*)d_in[2];  // [1536]
  const float* tbl     = (const float*)d_in[3];  // [169, 16]
  const float* proj_w  = (const float*)d_in[4];  // [512, 512]
  const float* proj_b  = (const float*)d_in[5];  // [512]
  float*       out     = (float*)d_out;          // [4096*49, 512]

  const int M = 4096 * 49;                       // 200704 (divisible by 128)
  const size_t qkv_bytes  = (size_t)M * 1536 * sizeof(u16);
  const size_t aout_bytes = (size_t)M * 512 * sizeof(u16);

  u16*   qkv       = (u16*)d_ws;
  u16*   aout      = (u16*)((char*)d_ws + qkv_bytes);
  float* bias_full = (float*)((char*)d_ws + qkv_bytes + aout_bytes);

  // 0) expand rel-pos bias into padded, mask-folded [16][64][64]
  build_bias_full<<<(16 * 64 * 64) / 256, 256, 0, stream>>>(tbl, bias_full);

  // 1) qkv = bf16( x @ qkv_w + qkv_b )
  dim3 g1(1536 / 128, M / 128);
  gemm_bf16_wmma<true, true><<<g1, 256, 0, stream>>>(
      x, qkv_w, qkv_b, qkv, M, 1536, 512);

  // 2) fused attention, one wave per (window, head): 4096*16 pairs
  const int pairs = 4096 * 16;
  window_attn_wmma<<<pairs / 4, 128, 0, stream>>>(qkv, bias_full, aout);

  // 3) out = aout @ proj_w + proj_b   (f32 output)
  dim3 g2(512 / 128, M / 128);
  gemm_bf16_wmma<false, false><<<g2, 256, 0, stream>>>(
      aout, proj_w, proj_b, out, M, 512, 512);
}
